// DiagonalBiLSTM_36326833389980
// MI455X (gfx1250) — matrix operand order repaired and checked by reading, not compiled, not run on hardware
//
#include <hip/hip_runtime.h>

// ---------------------------------------------------------------------------
// DiagonalBiLSTM on MI455X (gfx1250, wave32, WMMA)
//
// Pipeline:
//  K0 conv_weights : pack [Wi|Wh] (512x256) fwd/bwd and out_w (768x256) to bf16
//  K1 build_diag   : in_proj (3->128) + skew into per-diagonal step-major bf16
//  K2 lstm_scan    : per 16-row tile of (b,d) sequences, 32 sequential steps;
//                    gates = [x|h] @ [Wi|Wh]^T + b via v_wmma_f32_16x16x32_bf16;
//                    weights LDS-resident (256 KB of the 320 KB WGP LDS),
//                    cell state in registers, h recirculated through LDS A-tile.
//  K3 outproj      : un-skew gather + (256->768) WMMA GEMM + bias -> fp32 out
// ---------------------------------------------------------------------------

typedef __bf16 bf16;
typedef __attribute__((ext_vector_type(8)))  __bf16 bf16x8;
typedef __attribute__((ext_vector_type(16))) __bf16 v16bf;
typedef __attribute__((ext_vector_type(8)))  float  v8f;

#define NB   32      // batch
#define ND   94      // diagonals (H + SW - 1)
#define NS   32      // max steps per diagonal
#define HC   128     // hidden channels
#define NROWS (NB*ND)      // 3008 independent sequences
#define NTILES (NROWS/16)  // 188

__device__ __forceinline__ float sigm(float x) { return 1.0f / (1.0f + __expf(-x)); }

// A-fragment (16x32 bf16, M x K), row-major activation tile with leading dim 256.
// Lane L: m = L%16, half = L/16. elems e -> K = k0 + half*16 + e (contiguous).
__device__ __forceinline__ v16bf load_a_frag(const bf16* At, int m, int half, int k0) {
    const bf16* p = At + m * 256 + k0 + half * 16;
    bf16x8 lo = *(const bf16x8*)p;
    bf16x8 hi = *(const bf16x8*)(p + 8);
    return __builtin_shufflevector(lo, hi, 0,1,2,3,4,5,6,7,8,9,10,11,12,13,14,15);
}

// B-fragment (32x16 bf16, K x N) from weights stored (row = output n, col = K), ld=256.
// Lane L: n = n0 + L%16, elems e -> K = k0 + half*16 + e  (32 contiguous bytes).
__device__ __forceinline__ v16bf load_b_frag(const bf16* W, int n, int half, int k0) {
    const bf16* p = W + n * 256 + k0 + half * 16;
    bf16x8 lo = *(const bf16x8*)p;
    bf16x8 hi = *(const bf16x8*)(p + 8);
    return __builtin_shufflevector(lo, hi, 0,1,2,3,4,5,6,7,8,9,10,11,12,13,14,15);
}

// -------------------------------- K0 ---------------------------------------
__global__ __launch_bounds__(256) void conv_weights(
    const float* __restrict__ fWi, const float* __restrict__ fWh,
    const float* __restrict__ bWi, const float* __restrict__ bWh,
    const float* __restrict__ ow,
    bf16* __restrict__ WcatF, bf16* __restrict__ WcatB, bf16* __restrict__ Wo)
{
    int idx = blockIdx.x * 256 + threadIdx.x;
    if (idx < 512 * 256) {
        int n = idx >> 8, k = idx & 255;
        WcatF[idx] = (bf16)((k < 128) ? fWi[n * 128 + k] : fWh[n * 128 + (k - 128)]);
        WcatB[idx] = (bf16)((k < 128) ? bWi[n * 128 + k] : bWh[n * 128 + (k - 128)]);
    } else {
        int j = idx - 512 * 256;
        if (j < 768 * 256) Wo[j] = (bf16)ow[j];
    }
}

// -------------------------------- K1 ---------------------------------------
// diag layout: [row_g = b*94+d][s][128] bf16, zero outside the valid image strip.
__global__ __launch_bounds__(256) void build_diag(
    const float* __restrict__ x, const float* __restrict__ ipw,
    const float* __restrict__ ipb, bf16* __restrict__ diag)
{
    const int TOT = NROWS * NS * HC;
    int idx = blockIdx.x * 256 + threadIdx.x;
    if (idx >= TOT) return;
    int ch = idx & 127;
    int cell = idx >> 7;
    int s = cell & 31;
    int bd = cell >> 5;
    int d = bd % ND, b = bd / ND;
    int r0 = d > 62 ? d - 62 : 0;
    int mn = d < 31 ? d : 31;
    int len = mn - r0 + 1;
    int row = r0 + s;
    int cw = d - 2 * row;          // un-skewed column
    float val = 0.0f;
    if (s < len && cw >= 0 && cw < 32) {
        int base = b * 3 * 1024 + row * 32 + cw;
        float acc = ipb[ch];
        acc += x[base]        * (1.0f / 255.0f) * ipw[ch * 3 + 0];
        acc += x[base + 1024] * (1.0f / 255.0f) * ipw[ch * 3 + 1];
        acc += x[base + 2048] * (1.0f / 255.0f) * ipw[ch * 3 + 2];
        val = acc;
    }
    diag[idx] = (bf16)val;
}

// -------------------------------- K2 ---------------------------------------
// Grid: (188 tiles, 2 dirs). Block: 256 threads = 8 waves.
// Dynamic LDS: Wcat 512x256 bf16 (256 KB) + A-tile 16x256 bf16 (8 KB).
// Wave w owns N-tiles {w, w+8, w+16, w+24} => lane holds i,f,o,g of unit j=16w+lane%16.
__global__ __launch_bounds__(256) void lstm_scan(
    const bf16* __restrict__ diag,
    const bf16* __restrict__ WcatF, const bf16* __restrict__ WcatB,
    const float* __restrict__ biasF, const float* __restrict__ biasB,
    bf16* __restrict__ houtF, bf16* __restrict__ houtB)
{
    extern __shared__ char smem_raw[];
    bf16* sW = (bf16*)smem_raw;        // [512][256]
    bf16* sA = sW + 512 * 256;         // [16][256] : cols 0..127 = x(s), 128..255 = h

    const int tile = blockIdx.x;
    const int dir  = blockIdx.y;
    const bf16*  Wg   = dir ? WcatB : WcatF;
    const float* bias = dir ? biasB : biasF;
    bf16*        hout = dir ? houtB : houtF;

    const int tid  = threadIdx.x;
    const int lane = tid & 31, w = tid >> 5;
    const int half = lane >> 4, nsub = lane & 15;

    // Cooperative load of concatenated weights into LDS (b128 chunks).
    for (int i = tid; i < 512 * 256 / 8; i += 256)
        ((bf16x8*)sW)[i] = ((const bf16x8*)Wg)[i];

    // Zero h region of the A-tile.
    {
        int m = tid >> 4, seg = tid & 15;
        bf16* p = &sA[m * 256 + 128 + seg * 8];
#pragma unroll
        for (int e = 0; e < 8; ++e) p[e] = (bf16)0.0f;
    }

    // Per-lane row metadata (8 rows: m = half*8 + r) + cell state.
    float c[8];
    int len8[8], rowg8[8];
#pragma unroll
    for (int r = 0; r < 8; ++r) {
        c[r] = 0.0f;
        int rg = tile * 16 + half * 8 + r;
        rowg8[r] = rg;
        int d = rg % ND;
        int r0 = d > 62 ? d - 62 : 0;
        int mn = d < 31 ? d : 31;
        len8[r] = mn - r0 + 1;
    }
    float bias_r[4];
#pragma unroll
    for (int t = 0; t < 4; ++t) bias_r[t] = bias[(w + t * 8) * 16 + nsub];

    // Per-thread x-loader metadata (thread -> row m = tid>>4, 8-col chunk seg).
    const int xm = tid >> 4, xseg = tid & 15;
    const int xrg = tile * 16 + xm;
    int xd = xrg % ND;
    int xr0 = xd > 62 ? xd - 62 : 0;
    int xmn = xd < 31 ? xd : 31;
    const int xlen = xmn - xr0 + 1;

    for (int s = 0; s < NS; ++s) {
        // Stage x(s) (reversed per-row for backward direction).
        int src_s = (dir == 0) ? s : ((s < xlen) ? (xlen - 1 - s) : s);
        const bf16* src = diag + ((size_t)xrg * NS + src_s) * HC + xseg * 8;
        *(bf16x8*)&sA[xm * 256 + xseg * 8] = *(const bf16x8*)src;
        __syncthreads();

        // gates(16x512) = [x|h](16x256) @ Wcat^T + b  via 32 WMMA per wave.
        v8f acc[4];
#pragma unroll
        for (int t = 0; t < 4; ++t)
#pragma unroll
            for (int r = 0; r < 8; ++r) acc[t][r] = bias_r[t];

#pragma unroll
        for (int k = 0; k < 8; ++k) {
            v16bf a = load_a_frag(sA, nsub, half, k * 32);
#pragma unroll
            for (int t = 0; t < 4; ++t) {
                int n = (w + t * 8) * 16 + nsub;
                v16bf bfr = load_b_frag(sW, n, half, k * 32);
                acc[t] = __builtin_amdgcn_wmma_f32_16x16x32_bf16(
                    false, a, false, bfr, (short)0, acc[t], false, false);
            }
        }
        __syncthreads();   // protect A-tile h region until all reads complete

        // Elementwise LSTM update; lane owns hidden unit j across 8 rows.
        int j = w * 16 + nsub;
#pragma unroll
        for (int r = 0; r < 8; ++r) {
            if (s < len8[r]) {
                float i_ = sigm(acc[0][r]);
                float f_ = sigm(acc[1][r]);
                float o_ = sigm(acc[2][r]);
                float g_ = tanhf(acc[3][r]);
                float cn = f_ * c[r] + i_ * g_;
                c[r] = cn;
                float hn = o_ * tanhf(cn);
                bf16 hb = (bf16)hn;
                sA[(half * 8 + r) * 256 + 128 + j] = hb;                       // recirculate
                hout[((size_t)rowg8[r] * NS + s) * HC + j] = hb;               // emit
            }
        }
        // next iteration's x-stage + __syncthreads orders these h writes
    }
}

// -------------------------------- K3 ---------------------------------------
// logits(32768 x 768) = un(32768 x 256) @ out_w^T + out_b.
// Grid: 2048 row-tiles of 16 pixels; 8 waves x 6 N-tiles = 48 N-tiles (768 cols).
__global__ __launch_bounds__(256) void outproj(
    const bf16* __restrict__ houtF, const bf16* __restrict__ houtB,
    const bf16* __restrict__ Wo, const float* __restrict__ ob,
    float* __restrict__ out)
{
    __shared__ bf16 sA[16 * 256];
    const int tile = blockIdx.x;
    const int tid  = threadIdx.x;
    const int lane = tid & 31, w = tid >> 5;
    const int half = lane >> 4, nsub = lane & 15;

    // Gather: pixel (b,r,c) -> diagonal d = 2r+c, step s = r - r0(d); bwd is reversed.
    {
        int m = tid >> 4, seg = tid & 15;
        int flat = tile * 16 + m;
        int b = flat >> 10, rc = flat & 1023;
        int r = rc >> 5, cw = rc & 31;
        int d = 2 * r + cw;
        int r0 = d > 62 ? d - 62 : 0;
        int mn = d < 31 ? d : 31;
        int len = mn - r0 + 1;
        int s = r - r0;
        int rg = b * ND + d;
        int col0 = seg * 16;
        const bf16* src;
        if (col0 < 128) src = houtF + ((size_t)rg * NS + s) * HC + col0;
        else            src = houtB + ((size_t)rg * NS + (len - 1 - s)) * HC + (col0 - 128);
        *(bf16x8*)&sA[m * 256 + col0]     = *(const bf16x8*)src;
        *(bf16x8*)&sA[m * 256 + col0 + 8] = *(const bf16x8*)(src + 8);
    }
    __syncthreads();

    v8f acc[6];
    int ncol[6];
#pragma unroll
    for (int t = 0; t < 6; ++t) {
        ncol[t] = (w + t * 8) * 16 + nsub;
        float bv = ob[ncol[t]];
#pragma unroll
        for (int r = 0; r < 8; ++r) acc[t][r] = bv;
    }
#pragma unroll
    for (int k = 0; k < 16; ++k) {
        v16bf a = load_a_frag(sA, nsub, half, k * 32);
#pragma unroll
        for (int t = 0; t < 6; ++t) {
            v16bf bfr = load_b_frag(Wo, ncol[t], half, k * 32);
            acc[t] = __builtin_amdgcn_wmma_f32_16x16x32_bf16(
                false, a, false, bfr, (short)0, acc[t], false, false);
        }
    }
    // Scatter fp32: out[b][n][r][c]; tile rows map to consecutive (r*32+c).
#pragma unroll
    for (int t = 0; t < 6; ++t) {
        int n = ncol[t];
#pragma unroll
        for (int r = 0; r < 8; ++r) {
            int flat = tile * 16 + half * 8 + r;
            int b = flat >> 10, rc = flat & 1023;
            out[(size_t)b * (768 * 1024) + (size_t)n * 1024 + rc] = acc[t][r];
        }
    }
}

// ---------------------------------------------------------------------------
extern "C" void kernel_launch(void* const* d_in, const int* in_sizes, int n_in,
                              void* d_out, int out_size, void* d_ws, size_t ws_size,
                              hipStream_t stream)
{
    const float* x   = (const float*)d_in[0];
    const float* ipw = (const float*)d_in[1];
    const float* ipb = (const float*)d_in[2];
    const float* fWi = (const float*)d_in[3];
    const float* fWh = (const float*)d_in[4];
    const float* fb  = (const float*)d_in[5];
    const float* bWi = (const float*)d_in[6];
    const float* bWh = (const float*)d_in[7];
    const float* bb  = (const float*)d_in[8];
    const float* ow  = (const float*)d_in[9];
    const float* ob  = (const float*)d_in[10];
    float* out = (float*)d_out;

    char* ws = (char*)d_ws;
    size_t off = 0;
    auto salloc = [&](size_t bytes) {
        char* p = ws + off;
        off += (bytes + 255) & ~(size_t)255;
        return p;
    };
    bf16* diag  = (bf16*)salloc((size_t)NROWS * NS * HC * sizeof(bf16)); // 24.6 MB
    bf16* houtF = (bf16*)salloc((size_t)NROWS * NS * HC * sizeof(bf16)); // 24.6 MB
    bf16* houtB = (bf16*)salloc((size_t)NROWS * NS * HC * sizeof(bf16)); // 24.6 MB
    bf16* WcatF = (bf16*)salloc(512 * 256 * sizeof(bf16));
    bf16* WcatB = (bf16*)salloc(512 * 256 * sizeof(bf16));
    bf16* Wo    = (bf16*)salloc(768 * 256 * sizeof(bf16));

    conv_weights<<<1280, 256, 0, stream>>>(fWi, fWh, bWi, bWh, ow, WcatF, WcatB, Wo);
    build_diag<<<48128, 256, 0, stream>>>(x, ipw, ipb, diag);

    size_t smem = (size_t)(512 * 256 + 16 * 256) * sizeof(bf16); // 264 KB < 320 KB WGP LDS
    lstm_scan<<<dim3(NTILES, 2), 256, smem, stream>>>(diag, WcatF, WcatB, fb, bb, houtF, houtB);

    outproj<<<2048, 256, 0, stream>>>(houtF, houtB, Wo, ob, out);
}
